// MoE_49426483642525
// MI455X (gfx1250) — compile-verified, matrix operand order checked by reading, not run on hardware
//
#include <hip/hip_runtime.h>

// Problem constants (B,T,C = 4,4096,512 ; E=8, K=2, NS=2, H=4C)
#define NTOK 16384
#define BB   4
#define TT   4096
#define CC   512
#define EE   8
#define HH   2048
#define HCH  256   // H-chunk staged in LDS between GEMM1 and GEMM2

typedef __attribute__((ext_vector_type(16))) __bf16 v16bf;
typedef __attribute__((ext_vector_type(8)))  float  v8f;
typedef __attribute__((ext_vector_type(4)))  unsigned int u32x4;
typedef __attribute__((ext_vector_type(8)))  int  i32x8;
typedef __attribute__((ext_vector_type(4)))  int  i32x4;

union Frag16 { v16bf v; uint4 q[2]; };

// Load one 16-lane-wide WMMA operand fragment (A or B) from row-major bf16
// storage. Caller passes p = base + row*stride + k0 + ((lane>>4)<<3).
// Mapping per CDNA5 ISA 7.12.2 (16-bit A 16x32): elems 0..7 -> K+0..7,
// elems 8..15 -> K+16..23 (lanes 16..31 have K base +8).
__device__ __forceinline__ v16bf load_frag16(const __bf16* p) {
  Frag16 f;
  f.q[0] = *reinterpret_cast<const uint4*>(p);
  f.q[1] = *reinterpret_cast<const uint4*>(p + 16);
  return f.v;
}

__device__ __forceinline__ v8f wmma_bf16(v16bf a, v16bf b, v8f c) {
  return __builtin_amdgcn_wmma_f32_16x16x32_bf16(
      /*neg_a=*/false, a, /*neg_b=*/false, b,
      /*c_mod=*/(short)0, c, /*reuse_a=*/false, /*reuse_b=*/false);
}

__device__ __forceinline__ float gelu_tanh(float v) {
  const float k0 = 0.7978845608028654f;  // sqrt(2/pi)
  const float k1 = 0.044715f;
  float v3 = v * v * v;
  return 0.5f * v * (1.0f + tanhf(k0 * (v + k1 * v3)));
}

__device__ __forceinline__ v8f v8f_zero() {
  v8f z;
#pragma unroll
  for (int i = 0; i < 8; ++i) z[i] = 0.0f;
  return z;
}

// ---------------- Tensor Data Mover issue helper ---------------------------
// Descriptor layout per CDNA5 ISA ch.8 (D# groups 0..3). ROCm7.2 clang-22
// exposes the 5-arg builtin; clang-23 (therock) the 6-arg form.
__device__ __forceinline__ void tdm_load(u32x4 g0, i32x8 g1, i32x4 g2, i32x4 g3) {
#if __clang_major__ >= 23
  i32x8 gx;
#pragma unroll
  for (int i = 0; i < 8; ++i) gx[i] = 0;
  __builtin_amdgcn_tensor_load_to_lds(g0, g1, g2, g3, gx, 0);
#else
  __builtin_amdgcn_tensor_load_to_lds(g0, g1, g2, g3, 0);
#endif
}

// Gather-mode TDM: pull `nvalid` (<=16) rows of `rowlen` bf16 elements from
// a [nrows x rowlen] bf16 tensor at `gbase`, rows selected by 16-bit indices
// idx[0..7] (packed pairs), into LDS byte offset `lds_off`.
__device__ __forceinline__ void tdm_gather_rows(unsigned lds_off,
                                                const __bf16* gbase,
                                                unsigned rowlen, unsigned nrows,
                                                int nvalid, const int* idx8dw) {
  unsigned long long ga = (unsigned long long)(size_t)gbase;
  u32x4 g0;
  g0[0] = 0x80000001u;                       // gather_mode=1, idx16, count=1
  g0[1] = lds_off;                           // LDS byte address of tile
  g0[2] = (unsigned)(ga & 0xFFFFFFFFu);      // global_addr[31:0]
  g0[3] = (unsigned)((ga >> 32) & 0x01FFFFFFu) | (2u << 30);  // addr[56:32]|type=2
  i32x8 g1;
  g1[0] = 0x00010000;                        // data_size=1 (2B elements)
  g1[1] = (int)(rowlen << 16);               // tensor_dim0 (lo16 in hi half)
  g1[2] = (int)(nrows << 16);                // tensor_dim1
  g1[3] = (int)(rowlen << 16);               // tile_dim0
  g1[4] = nvalid;                            // tile_dim1 = #valid indices
  g1[5] = (int)rowlen;                       // tensor_dim0_stride
  g1[6] = 0;
  g1[7] = 0;
  i32x4 g2, g3;
#pragma unroll
  for (int i = 0; i < 4; ++i) { g2[i] = idx8dw[i]; g3[i] = idx8dw[4 + i]; }
  tdm_load(g0, g1, g2, g3);
}

// Plain 2D TDM tile load: [tile_rows x rowlen] bf16 tile starting at gbase.
__device__ __forceinline__ void tdm_tile2d(unsigned lds_off,
                                           const __bf16* gbase,
                                           unsigned rowlen, unsigned rows_left,
                                           unsigned tile_rows) {
  unsigned long long ga = (unsigned long long)(size_t)gbase;
  u32x4 g0;
  g0[0] = 0x00000001u;                       // count=1
  g0[1] = lds_off;
  g0[2] = (unsigned)(ga & 0xFFFFFFFFu);
  g0[3] = (unsigned)((ga >> 32) & 0x01FFFFFFu) | (2u << 30);
  i32x8 g1;
  g1[0] = 0x00010000;                        // data_size=1 (2B)
  g1[1] = (int)(rowlen << 16);               // tensor_dim0
  g1[2] = (int)((rows_left & 0xFFFFu) << 16);// tensor_dim1 lo16
  g1[3] = (int)(((rows_left >> 16) & 0xFFFFu) | (rowlen << 16)); // dim1 hi | tile_dim0
  g1[4] = (int)tile_rows;                    // tile_dim1 (tile_dim2=0)
  g1[5] = (int)rowlen;                       // tensor_dim0_stride
  g1[6] = 0;
  g1[7] = 0;
  i32x4 gz;
#pragma unroll
  for (int i = 0; i < 4; ++i) gz[i] = 0;
  tdm_load(g0, g1, gz, gz);
}

// ---------------- init: zero expert counters and f/p stats -----------------
__global__ void moe_init_kernel(int* __restrict__ cnt, float* __restrict__ fp) {
  int t = threadIdx.x;
  if (t < EE) cnt[t] = 0;
  if (t < 2 * BB * EE) fp[t] = 0.0f;
}

// ---------------- fp32 -> bf16 conversion of activations ------------------
__global__ void moe_cvt_x_kernel(const float* __restrict__ x,
                                 __bf16* __restrict__ xb) {
  int i = blockIdx.x * blockDim.x + threadIdx.x;   // i < NTOK*CC/4
  float4 v = reinterpret_cast<const float4*>(x)[i];
  __bf16* o = xb + (size_t)i * 4;
  o[0] = (__bf16)v.x; o[1] = (__bf16)v.y;
  o[2] = (__bf16)v.z; o[3] = (__bf16)v.w;
}

// Fold shared-expert weight over NS heads and transpose: WsT[n][c] (bf16)
__global__ void moe_fold_ws_kernel(const float* __restrict__ Ws,
                                   __bf16* __restrict__ WsT) {
  int i = blockIdx.x * blockDim.x + threadIdx.x;   // i < CC*CC
  int c = i % CC;
  int n = i / CC;
  float v = Ws[(size_t)c * (2 * CC) + n] + Ws[(size_t)c * (2 * CC) + CC + n];
  WsT[i] = (__bf16)v;
}

// W1: [E][C][H] -> W1T: [E][H][C] bf16   (B operand reads contiguous along K=C)
__global__ void moe_cvt_w1t_kernel(const float* __restrict__ W1,
                                   __bf16* __restrict__ W1T) {
  size_t i = (size_t)blockIdx.x * blockDim.x + threadIdx.x;  // < E*H*C
  int c = (int)(i % CC);
  size_t r = i / CC;
  int h = (int)(r % HH);
  int e = (int)(r / HH);
  W1T[i] = (__bf16)W1[((size_t)e * CC + c) * HH + h];
}

// W2: [E][H][C] -> W2T: [E][C][H] bf16   (B operand reads contiguous along K=H)
__global__ void moe_cvt_w2t_kernel(const float* __restrict__ W2,
                                   __bf16* __restrict__ W2T) {
  size_t i = (size_t)blockIdx.x * blockDim.x + threadIdx.x;  // < E*C*H
  int h = (int)(i % HH);
  size_t r = i / HH;
  int c = (int)(r % CC);
  int e = (int)(r / CC);
  W2T[i] = (__bf16)W2[((size_t)e * HH + h) * CC + c];
}

// ---------------- gate: sigmoid scores, top-2, routing lists, stats --------
__global__ void moe_gate_kernel(const float* __restrict__ x,
                                const float* __restrict__ Wg,
                                const float* __restrict__ bg,
                                int* __restrict__ cnt,
                                unsigned short* __restrict__ tokl16,
                                float* __restrict__ wtl,
                                float* __restrict__ f_out,
                                float* __restrict__ p_out) {
  int tok  = (blockIdx.x * blockDim.x + threadIdx.x) >> 5;  // wave per token
  int lane = threadIdx.x & 31;

  const float* xr = x + (size_t)tok * CC;
  float acc[EE];
#pragma unroll
  for (int e = 0; e < EE; ++e) acc[e] = 0.0f;

  for (int c = lane; c < CC; c += 32) {
    float xv = xr[c];
    const float* wr = Wg + (size_t)c * EE;
#pragma unroll
    for (int e = 0; e < EE; ++e) acc[e] += xv * wr[e];
  }
#pragma unroll
  for (int e = 0; e < EE; ++e) {
    float v = acc[e];
#pragma unroll
    for (int off = 16; off > 0; off >>= 1) v += __shfl_xor(v, off, 32);
    acc[e] = v;
  }

  if (lane == 0) {
    float ssum = 0.0f;
    float best0 = -1.0f, best1 = -1.0f;
    int i0 = 0, i1 = 0;
#pragma unroll
    for (int e = 0; e < EE; ++e) {
      float se = 1.0f / (1.0f + expf(-(acc[e] + bg[e])));
      ssum += se;
      if (se > best0) { best1 = best0; i1 = i0; best0 = se; i0 = e; }
      else if (se > best1) { best1 = se; i1 = e; }
    }
    float wsum = best0 + best1;
    int b = tok / TT;

    int p0 = atomicAdd(&cnt[i0], 1);
    tokl16[i0 * NTOK + p0] = (unsigned short)tok;
    wtl   [i0 * NTOK + p0] = best0 / wsum;
    int p1 = atomicAdd(&cnt[i1], 1);
    tokl16[i1 * NTOK + p1] = (unsigned short)tok;
    wtl   [i1 * NTOK + p1] = best1 / wsum;

    float sn0 = best0 / ssum, sn1 = best1 / ssum;
#pragma unroll
    for (int e = 0; e < EE; ++e) {
      float fc = (float)((i0 != e) ? 1 : 0) + (float)((i1 != e) ? 1 : 0);
      float pc = ((i0 != e) ? sn0 : 0.0f) + ((i1 != e) ? sn1 : 0.0f);
      atomicAdd(&f_out[b * EE + e], fc);
      atomicAdd(&p_out[b * EE + e], pc);
    }
  }
}

// ---------------- shared expert GEMM: res = x + xb @ WsT^T + bsum ----------
// Block: 256 threads (8 waves), 32-token tile, output [32 x 512].
// Activation tile staged into LDS by the Tensor Data Mover.
__global__ void moe_shared_gemm_kernel(const float* __restrict__ x,
                                       const __bf16* __restrict__ xb,
                                       const __bf16* __restrict__ WsT,
                                       const float* __restrict__ bs,
                                       float* __restrict__ res) {
  extern __shared__ __bf16 xg[];   // [32][CC]
  int tile0 = blockIdx.x * 32;

  if (threadIdx.x < 32) {  // wave 0 drives the TDM
    tdm_tile2d(/*lds_off=*/0u, xb + (size_t)tile0 * CC,
               /*rowlen=*/CC, /*rows_left=*/(unsigned)(NTOK - tile0),
               /*tile_rows=*/32u);
    __builtin_amdgcn_s_wait_tensorcnt(0);
  }
  __syncthreads();

  int w = threadIdx.x >> 5, lane = threadIdx.x & 31;
  int col = lane & 15, rsel = lane >> 4;
  int mt = w >> 2;                       // row-tile 0..1
  int ntb = (w & 3) * 8;                 // 8 col-tiles per wave

  v8f acc[8];
#pragma unroll
  for (int j = 0; j < 8; ++j) acc[j] = v8f_zero();

  for (int k0 = 0; k0 < CC; k0 += 32) {
    v16bf A = load_frag16(xg + (mt * 16 + col) * CC + k0 + rsel * 8);
#pragma unroll
    for (int j = 0; j < 8; ++j) {
      int n = (ntb + j) * 16 + col;
      v16bf Bf = load_frag16(WsT + (size_t)n * CC + k0 + rsel * 8);
      acc[j] = wmma_bf16(A, Bf, acc[j]);
    }
  }

  int rb = mt * 16 + rsel * 8;
#pragma unroll
  for (int j = 0; j < 8; ++j) {
    int n = (ntb + j) * 16 + col;
    float bsum = bs[n] + bs[CC + n];
#pragma unroll
    for (int i = 0; i < 8; ++i) {
      size_t o = (size_t)(tile0 + rb + i) * CC + n;
      res[o] = acc[j][i] + bsum + x[o];
    }
  }
}

// ---------------- routed expert FFN -----------------------------------------
// grid = (NTOK/32, EE). TDM gather-mode pulls 32 routed token rows into LDS,
// h = gelu(xg@W1+b1) chunked through LDS, y = h@W2+b2, res += w*y (atomic).
__global__ void moe_expert_ffn_kernel(const __bf16* __restrict__ xb,
                                      const __bf16* __restrict__ W1T,
                                      const __bf16* __restrict__ W2T,
                                      const float* __restrict__ b1,
                                      const float* __restrict__ b2,
                                      const int* __restrict__ cnt,
                                      const unsigned short* __restrict__ tokl16,
                                      const float* __restrict__ wtl,
                                      float* __restrict__ res) {
  int e = blockIdx.y;
  int cnt_e = cnt[e];
  int tile0 = blockIdx.x * 32;
  if (tile0 >= cnt_e) return;           // block-uniform early exit

  extern __shared__ char smraw[];
  __bf16* xg   = reinterpret_cast<__bf16*>(smraw);        // [32][CC]
  __bf16* hch  = xg + 32 * CC;                            // [32][HCH]
  float*  wrow = reinterpret_cast<float*>(hch + 32 * HCH);// [32]
  int*    trow = reinterpret_cast<int*>(wrow + 32);       // [32]

  int valid = cnt_e - tile0; if (valid > 32) valid = 32;

  if (threadIdx.x < 32) {
    int slot = tile0 + threadIdx.x;
    trow[threadIdx.x] = (slot < cnt_e) ? (int)tokl16[e * NTOK + slot] : -1;
    wrow[threadIdx.x] = (slot < cnt_e) ? wtl[e * NTOK + slot] : 0.0f;
  }
  {  // zero rows past the valid count (ragged last tile); disjoint from TDM
    int t = threadIdx.x, row = t >> 3, lr = t & 7;
    if (row >= valid) {
      uint4 z; z.x = z.y = z.z = z.w = 0u;
      uint4* dst = reinterpret_cast<uint4*>(xg + row * CC + lr * 64);
#pragma unroll
      for (int i = 0; i < 8; ++i) dst[i] = z;
    }
  }
  if (threadIdx.x < 32) {  // wave 0: TDM gather of routed token rows
    const int* idx = reinterpret_cast<const int*>(tokl16 + e * NTOK + tile0);
    int nv0 = (valid < 16) ? valid : 16;
    tdm_gather_rows(/*lds_off=*/0u, xb, CC, NTOK, nv0, idx);
    int nv1 = valid - 16;
    if (nv1 > 0)
      tdm_gather_rows(/*lds_off=*/16u * CC * 2u, xb, CC, NTOK, nv1, idx + 8);
    __builtin_amdgcn_s_wait_tensorcnt(0);
  }
  __syncthreads();

  int w = threadIdx.x >> 5, lane = threadIdx.x & 31;
  int col = lane & 15, rsel = lane >> 4;
  int mt = w >> 2;
  const __bf16* W1e = W1T + (size_t)e * HH * CC;
  const __bf16* W2e = W2T + (size_t)e * CC * HH;

  v8f yacc[8];
#pragma unroll
  for (int j = 0; j < 8; ++j) yacc[j] = v8f_zero();

  for (int ch = 0; ch < HH / HCH; ++ch) {
    int h0 = ch * HCH;

    if (lane == 0 && ch + 1 < HH / HCH) {
      // warm L2/WGP$ for next chunk's weight panels (global_prefetch_b8)
      __builtin_prefetch(W1e + (size_t)(h0 + HCH + (w & 3) * 64) * CC, 0, 3);
      __builtin_prefetch(W2e + (size_t)((w & 3) * 128 + 64) * HH + h0 + HCH, 0, 3);
    }

    // GEMM1 slice: [32 x HCH] = xg[32 x CC] @ W1[CC x HCH], 4 tiles/wave
    v8f hacc[4];
#pragma unroll
    for (int j = 0; j < 4; ++j) hacc[j] = v8f_zero();

    for (int k0 = 0; k0 < CC; k0 += 32) {
      v16bf A = load_frag16(xg + (mt * 16 + col) * CC + k0 + rsel * 8);
#pragma unroll
      for (int j = 0; j < 4; ++j) {
        int n = h0 + ((w & 3) * 4 + j) * 16 + col;
        v16bf Bf = load_frag16(W1e + (size_t)n * CC + k0 + rsel * 8);
        hacc[j] = wmma_bf16(A, Bf, hacc[j]);
      }
    }
    // bias + gelu -> bf16 LDS chunk
    int rb = mt * 16 + rsel * 8;
#pragma unroll
    for (int j = 0; j < 4; ++j) {
      int ncl = ((w & 3) * 4 + j) * 16 + col;     // column within chunk
      float bv = b1[(size_t)e * HH + h0 + ncl];
#pragma unroll
      for (int i = 0; i < 8; ++i) {
        float g = gelu_tanh(hacc[j][i] + bv);
        hch[(rb + i) * HCH + ncl] = (__bf16)g;
      }
    }
    __syncthreads();

    // GEMM2 partial: yacc += hch[32 x HCH] @ W2[HCH x CC], 8 tiles/wave
    for (int kk = 0; kk < HCH; kk += 32) {
      v16bf A = load_frag16(hch + (mt * 16 + col) * HCH + kk + rsel * 8);
#pragma unroll
      for (int j = 0; j < 8; ++j) {
        int n = ((w & 3) * 8 + j) * 16 + col;
        v16bf Bf = load_frag16(W2e + (size_t)n * HH + h0 + kk + rsel * 8);
        yacc[j] = wmma_bf16(A, Bf, yacc[j]);
      }
    }
    __syncthreads();   // protect hch before next chunk's GEMM1 overwrites it
  }

  // scatter-add gated output rows
  int rb = mt * 16 + rsel * 8;
#pragma unroll
  for (int j = 0; j < 8; ++j) {
    int n = ((w & 3) * 8 + j) * 16 + col;
    float bv = b2[(size_t)e * CC + n];
#pragma unroll
    for (int i = 0; i < 8; ++i) {
      int row = rb + i;
      int tok = trow[row];
      if (tok >= 0)
        atomicAdd(&res[(size_t)tok * CC + n], wrow[row] * (yacc[j][i] + bv));
    }
  }
}

// ---------------------------------------------------------------------------
extern "C" void kernel_launch(void* const* d_in, const int* in_sizes, int n_in,
                              void* d_out, int out_size, void* d_ws, size_t ws_size,
                              hipStream_t stream) {
  const float* x  = (const float*)d_in[0];
  const float* Wg = (const float*)d_in[1];
  const float* bg = (const float*)d_in[2];
  const float* Ws = (const float*)d_in[3];
  const float* bs = (const float*)d_in[4];
  const float* W1 = (const float*)d_in[5];
  const float* b1 = (const float*)d_in[6];
  const float* W2 = (const float*)d_in[7];
  const float* b2 = (const float*)d_in[8];

  float* res = (float*)d_out;                    // [NTOK, CC]
  float* fp  = res + (size_t)NTOK * CC;          // f[BB,EE] then p[BB,EE]

  // Workspace carve-up (bytes, 256-aligned)
  char* ws = (char*)d_ws;
  constexpr size_t off_xb  = 0;                                   // N*C bf16
  constexpr size_t off_wst = off_xb  + (size_t)NTOK * CC * 2;     // C*C bf16
  constexpr size_t off_w1t = off_wst + (size_t)CC * CC * 2;       // E*H*C bf16
  constexpr size_t off_w2t = off_w1t + (size_t)EE * HH * CC * 2;  // E*C*H bf16
  constexpr size_t off_cnt = off_w2t + (size_t)EE * CC * HH * 2;  // E ints
  constexpr size_t off_tok = off_cnt + 256;                       // E*N u16
  constexpr size_t off_wt  = off_tok + (size_t)EE * NTOK * 2;     // E*N floats

  __bf16* xb  = (__bf16*)(ws + off_xb);
  __bf16* WsT = (__bf16*)(ws + off_wst);
  __bf16* W1T = (__bf16*)(ws + off_w1t);
  __bf16* W2T = (__bf16*)(ws + off_w2t);
  int*    cnt = (int*)   (ws + off_cnt);
  unsigned short* tok = (unsigned short*)(ws + off_tok);
  float*  wtl = (float*) (ws + off_wt);

  moe_init_kernel<<<1, 64, 0, stream>>>(cnt, fp);

  moe_cvt_x_kernel<<<(NTOK * CC / 4) / 256, 256, 0, stream>>>(x, xb);
  moe_fold_ws_kernel<<<(CC * CC) / 256, 256, 0, stream>>>(Ws, WsT);
  moe_cvt_w1t_kernel<<<(EE * HH * CC) / 256, 256, 0, stream>>>(W1, W1T);
  moe_cvt_w2t_kernel<<<(EE * CC * HH) / 256, 256, 0, stream>>>(W2, W2T);

  moe_gate_kernel<<<(NTOK * 32) / 256, 256, 0, stream>>>(
      x, Wg, bg, cnt, tok, wtl, fp, fp + BB * EE);

  size_t smem_shared = (size_t)32 * CC * 2;   // 32 KB
  moe_shared_gemm_kernel<<<NTOK / 32, 256, smem_shared, stream>>>(
      x, xb, WsT, bs, res);

  size_t smem_ffn = (size_t)32 * CC * 2 + (size_t)32 * HCH * 2 + 32 * 4 + 32 * 4;
  dim3 ffn_grid(NTOK / 32, EE);
  moe_expert_ffn_kernel<<<ffn_grid, 256, smem_ffn, stream>>>(
      xb, W1T, W2T, b1, b2, cnt, tok, wtl, res);
}
